// NonLocalBlock_57062935494897
// MI455X (gfx1250) — compile-verified, matrix operand order checked by reading, not run on hardware
//
#include <hip/hip_runtime.h>
#include <stdint.h>

typedef __attribute__((ext_vector_type(16))) __bf16 v16bf;
typedef __attribute__((ext_vector_type(8)))  float  v8f;

#define Bn 4
#define Cc 256
#define Dd 128
#define Nn 4096
#define EPSV 1e-5f

__device__ __forceinline__ uint16_t f2bf(float f) {
  uint32_t u = __float_as_uint(f);
  u += 0x7FFFu + ((u >> 16) & 1u);   // round-to-nearest-even
  return (uint16_t)(u >> 16);
}

// async 16B global -> LDS copy (per lane), tracked by ASYNCcnt
__device__ __forceinline__ void async_cp16(const uint16_t* g, uint16_t* l) {
  asm volatile("global_load_async_to_lds_b128 %0, %1, off"
               :: "v"((uint32_t)(size_t)l), "v"((uint64_t)(size_t)g)
               : "memory");
}

// ---------------- elementwise f32 -> bf16 convert (weights) ----------------
__global__ void cvt_bf16_kernel(const float* __restrict__ in, uint16_t* __restrict__ out, int n) {
  int i = blockIdx.x * blockDim.x + threadIdx.x;
  if (i < n) out[i] = f2bf(in[i]);
}

// ---------------- x [B][C][N] f32 -> xt [B][N][C] bf16 ----------------
__global__ void transpose_cvt_x(const float* __restrict__ x, uint16_t* __restrict__ xt) {
  __shared__ float t[32][33];
  const int b = blockIdx.z, c0 = blockIdx.y * 32, n0 = blockIdx.x * 32;
  const int tx = threadIdx.x, ty = threadIdx.y;
#pragma unroll
  for (int i = 0; i < 4; ++i)
    t[ty + 8 * i][tx] = x[((size_t)b * Cc + c0 + ty + 8 * i) * Nn + n0 + tx];
  __syncthreads();
#pragma unroll
  for (int i = 0; i < 4; ++i)
    xt[((size_t)b * Nn + n0 + ty + 8 * i) * Cc + c0 + tx] = f2bf(t[tx][ty + 8 * i]);
}

// ---------------- gy [B][N][D] bf16 -> gT [B][D][N] bf16 ----------------
__global__ void transpose_g(const uint16_t* __restrict__ gy, uint16_t* __restrict__ gT) {
  __shared__ uint16_t t[32][33];
  const int b = blockIdx.z, n0 = blockIdx.y * 32, d0 = blockIdx.x * 32;
  const int tx = threadIdx.x, ty = threadIdx.y;
#pragma unroll
  for (int i = 0; i < 4; ++i)
    t[ty + 8 * i][tx] = gy[((size_t)b * Nn + n0 + ty + 8 * i) * Dd + d0 + tx];
  __syncthreads();
#pragma unroll
  for (int i = 0; i < 4; ++i)
    gT[((size_t)b * Dd + d0 + ty + 8 * i) * Nn + n0 + tx] = t[tx][ty + 8 * i];
}

// ---------------- projection GEMM: out[b][n][d] = (xt[b][n][:] . w[d][:] + bias[d]) * scale
// grid.x = B*(N/16); block 256 = 8 waves, wave -> d-tile
__global__ __launch_bounds__(256) void proj_gemm_kernel(
    const uint16_t* __restrict__ xt, const uint16_t* __restrict__ w,
    const float* __restrict__ bias, float scale, uint16_t* __restrict__ out) {
  const int wave = threadIdx.x >> 5, lane = threadIdx.x & 31;
  const int ln = lane & 15, hk = lane >> 4;
  const int b = blockIdx.x >> 8;
  const int q0 = (blockIdx.x & 255) << 4;
  const int d0 = wave << 4;

  const uint16_t* xrow = xt + ((size_t)b * Nn + q0 + ln) * Cc;
  const uint16_t* wrow = w + (size_t)(d0 + ln) * Cc;

  v8f acc = {};
#pragma unroll
  for (int kk = 0; kk < 8; ++kk) {
    v16bf a  = *reinterpret_cast<const v16bf*>(xrow + kk * 32 + hk * 16);
    v16bf bm = *reinterpret_cast<const v16bf*>(wrow + kk * 32 + hk * 16);
    acc = __builtin_amdgcn_wmma_f32_16x16x32_bf16(false, a, false, bm, (short)0, acc, false, false);
  }
  const float bv = bias[d0 + ln];
#pragma unroll
  for (int r = 0; r < 8; ++r) {
    float v = (acc[r] + bv) * scale;
    out[((size_t)b * Nn + q0 + r + 8 * hk) * Dd + d0 + ln] = f2bf(v);
  }
}

// ---------------- flash attention with async-LDS K/V staging ----------------
// block = 128 threads = 4 waves (one 16-query tile each, 64 queries/block),
// K/V chunks shared across the block via GLOBAL_LOAD_ASYNC_TO_LDS double buffering.
__global__ __launch_bounds__(128) void flash_attn_kernel(
    const uint16_t* __restrict__ th, const uint16_t* __restrict__ ph,
    const uint16_t* __restrict__ gT, uint16_t* __restrict__ ybf) {
  const int tid = threadIdx.x;
  const int wave = tid >> 5, lane = tid & 31;
  const int ln = lane & 15, hk = lane >> 4;
  const int b = blockIdx.x >> 6;                  // 64 blocks per batch
  const int q0 = (blockIdx.x & 63) * 64 + wave * 16;

  const uint16_t* thB = th + (size_t)b * Nn * Dd;
  const uint16_t* phB = ph + (size_t)b * Nn * Dd;
  const uint16_t* gB  = gT + (size_t)b * Dd * Nn;

  __shared__ __align__(32) uint16_t kbuf[2][32 * 128];   // K chunk: 32 keys x 128 d
  __shared__ __align__(32) uint16_t vbuf[2][128 * 32];   // V chunk: 128 d x 32 keys
  __shared__ __align__(32) uint16_t pstage[4][16 * 32];  // per-wave P staging

  // issue one chunk's async copies: 4+4 instrs per wave, 16B per lane each
  auto issue_chunk = [&](int k0, int bufi) {
    const uint16_t* ksrc = phB + (size_t)k0 * Dd;  // 8KB contiguous
#pragma unroll
    for (int t = 0; t < 4; ++t) {
      const int j = t * 128 + tid;
      async_cp16(ksrc + (size_t)j * 8, &kbuf[bufi][j * 8]);
    }
#pragma unroll
    for (int t = 0; t < 4; ++t) {
      const int j = t * 128 + tid;
      const int d = j >> 2, seg = j & 3;             // 128 d-rows x 4 x 16B segments
      async_cp16(gB + (size_t)d * Nn + k0 + seg * 8, &vbuf[bufi][d * 32 + seg * 8]);
    }
  };

  v16bf qa[4];
#pragma unroll
  for (int kk = 0; kk < 4; ++kk)
    qa[kk] = *reinterpret_cast<const v16bf*>(thB + (size_t)(q0 + ln) * Dd + kk * 32 + hk * 16);

  v8f o[8];
#pragma unroll
  for (int t = 0; t < 8; ++t) o[t] = (v8f){};
  float rm[8], rs[8];
#pragma unroll
  for (int r = 0; r < 8; ++r) { rm[r] = -3.0e38f; rs[r] = 0.0f; }

  issue_chunk(0, 0);

  for (int it = 0; it < Nn / 32; ++it) {
    const int cur = it & 1;
    const int k0 = it * 32;
    if (k0 + 32 < Nn) {
      issue_chunk(k0 + 32, cur ^ 1);
      asm volatile("s_wait_asynccnt 0x8" ::: "memory");  // oldest 8 (current chunk) done
    } else {
      asm volatile("s_wait_asynccnt 0x0" ::: "memory");
    }
    __syncthreads();

    // ---- S = Q K^T (two 16x16 key tiles), operands from LDS ----
    v8f s0 = {}, s1 = {};
#pragma unroll
    for (int kk = 0; kk < 4; ++kk) {
      v16bf kb0 = *reinterpret_cast<const v16bf*>(&kbuf[cur][ln * 128 + kk * 32 + hk * 16]);
      v16bf kb1 = *reinterpret_cast<const v16bf*>(&kbuf[cur][(16 + ln) * 128 + kk * 32 + hk * 16]);
      s0 = __builtin_amdgcn_wmma_f32_16x16x32_bf16(false, qa[kk], false, kb0, (short)0, s0, false, false);
      s1 = __builtin_amdgcn_wmma_f32_16x16x32_bf16(false, qa[kk], false, kb1, (short)0, s1, false, false);
    }

    // ---- online softmax (C-layout rows: lane half hk owns rows r+8*hk) ----
    float alpha[8], p0[8], p1[8];
#pragma unroll
    for (int r = 0; r < 8; ++r) {
      float v = fmaxf(s0[r], s1[r]);
      v = fmaxf(v, __shfl_xor(v, 1));
      v = fmaxf(v, __shfl_xor(v, 2));
      v = fmaxf(v, __shfl_xor(v, 4));
      v = fmaxf(v, __shfl_xor(v, 8));
      const float mn = fmaxf(rm[r], v);
      alpha[r] = __expf(rm[r] - mn);
      p0[r] = __expf(s0[r] - mn);
      p1[r] = __expf(s1[r] - mn);
      float ps = p0[r] + p1[r];
      ps += __shfl_xor(ps, 1);
      ps += __shfl_xor(ps, 2);
      ps += __shfl_xor(ps, 4);
      ps += __shfl_xor(ps, 8);
      rs[r] = rs[r] * alpha[r] + ps;
      rm[r] = mn;
    }
#pragma unroll
    for (int t = 0; t < 8; ++t)
#pragma unroll
      for (int r = 0; r < 8; ++r) o[t][r] *= alpha[r];

    // ---- C-layout -> A-layout for P via wave-private LDS tile ----
    uint16_t* ltile = pstage[wave];
#pragma unroll
    for (int r = 0; r < 8; ++r) {
      ltile[(r + 8 * hk) * 32 + ln]      = f2bf(p0[r]);
      ltile[(r + 8 * hk) * 32 + 16 + ln] = f2bf(p1[r]);
    }
    asm volatile("s_wait_dscnt 0x0" ::: "memory");
    v16bf pa = *reinterpret_cast<const v16bf*>(ltile + ln * 32 + hk * 16);

    // ---- O += P V, V tiles from LDS ----
#pragma unroll
    for (int t = 0; t < 8; ++t) {
      v16bf vb = *reinterpret_cast<const v16bf*>(&vbuf[cur][(t * 16 + ln) * 32 + hk * 16]);
      o[t] = __builtin_amdgcn_wmma_f32_16x16x32_bf16(false, pa, false, vb, (short)0, o[t], false, false);
    }
    __syncthreads();   // all waves done reading buf[cur] before it is overwritten
  }

#pragma unroll
  for (int r = 0; r < 8; ++r) {
    const float inv = 1.0f / rs[r];
    const int q = q0 + r + 8 * hk;
#pragma unroll
    for (int t = 0; t < 8; ++t)
      ybf[((size_t)b * Nn + q) * Dd + t * 16 + ln] = f2bf(o[t][r] * inv);
  }
}

// ---------------- out conv GEMM + BN statistics ----------------
// grid (B*N/16, 2), block 256: wave -> c-tile
__global__ __launch_bounds__(256) void outconv_kernel(
    const uint16_t* __restrict__ ybf, const uint16_t* __restrict__ wout,
    const float* __restrict__ ob, float* __restrict__ zpre,
    float* __restrict__ bnsum, float* __restrict__ bnsq) {
  const int wave = threadIdx.x >> 5, lane = threadIdx.x & 31;
  const int ln = lane & 15, hk = lane >> 4;
  const int b = blockIdx.x >> 8;
  const int n0 = (blockIdx.x & 255) << 4;
  const int c0 = (blockIdx.y * 8 + wave) << 4;

  const uint16_t* yrow = ybf + ((size_t)b * Nn + n0 + ln) * Dd;
  const uint16_t* wrow = wout + (size_t)(c0 + ln) * Dd;
  v8f acc = {};
#pragma unroll
  for (int kk = 0; kk < 4; ++kk) {
    v16bf a  = *reinterpret_cast<const v16bf*>(yrow + kk * 32 + hk * 16);
    v16bf bm = *reinterpret_cast<const v16bf*>(wrow + kk * 32 + hk * 16);
    acc = __builtin_amdgcn_wmma_f32_16x16x32_bf16(false, a, false, bm, (short)0, acc, false, false);
  }
  const int c = c0 + ln;
  const float bv = ob[c];
  float s = 0.f, sq = 0.f;
#pragma unroll
  for (int r = 0; r < 8; ++r) {
    float z = acc[r] + bv;
    zpre[((size_t)b * Cc + c) * Nn + n0 + r + 8 * hk] = z;
    s += z; sq += z * z;
  }
  atomicAdd(&bnsum[c], s);
  atomicAdd(&bnsq[c], sq);
}

__global__ void init_stats(float* __restrict__ a, float* __restrict__ b2) {
  int i = threadIdx.x;
  if (i < Cc) { a[i] = 0.0f; b2[i] = 0.0f; }
}

// ---------------- BN (batch stats) + residual ----------------
__global__ __launch_bounds__(256) void bn_residual_kernel(
    const float* __restrict__ x, const float* __restrict__ zpre,
    const float* __restrict__ bnsum, const float* __restrict__ bnsq,
    const float* __restrict__ gamma, const float* __restrict__ beta,
    float* __restrict__ out) {
  size_t i = (size_t)blockIdx.x * blockDim.x + threadIdx.x;
  const int c = (int)((i / Nn) % Cc);
  const float cnt = (float)(Bn * Nn);
  const float mean = bnsum[c] / cnt;
  const float var = bnsq[c] / cnt - mean * mean;
  out[i] = x[i] + gamma[c] * (zpre[i] - mean) * rsqrtf(var + EPSV) + beta[c];
}

extern "C" void kernel_launch(void* const* d_in, const int* in_sizes, int n_in,
                              void* d_out, int out_size, void* d_ws, size_t ws_size,
                              hipStream_t stream) {
  const float* x       = (const float*)d_in[0];
  const float* theta_w = (const float*)d_in[1];
  const float* theta_b = (const float*)d_in[2];
  const float* phi_w   = (const float*)d_in[3];
  const float* phi_b   = (const float*)d_in[4];
  const float* g_w     = (const float*)d_in[5];
  const float* g_b     = (const float*)d_in[6];
  const float* out_w   = (const float*)d_in[7];
  const float* out_b   = (const float*)d_in[8];
  const float* gamma   = (const float*)d_in[9];
  const float* beta    = (const float*)d_in[10];
  float* out = (float*)d_out;

  char* ws = (char*)d_ws;
  size_t off = 0;
  auto alloc = [&](size_t bytes) -> void* {
    void* p = ws + off;
    off += (bytes + 255) & ~(size_t)255;
    return p;
  };
  uint16_t* xt    = (uint16_t*)alloc((size_t)Bn * Nn * Cc * 2);  // x^T bf16
  uint16_t* thbf  = (uint16_t*)alloc((size_t)Bn * Nn * Dd * 2);  // theta [B][N][D]
  uint16_t* phbf  = (uint16_t*)alloc((size_t)Bn * Nn * Dd * 2);  // phi   [B][N][D]
  uint16_t* gybf  = (uint16_t*)alloc((size_t)Bn * Nn * Dd * 2);  // g     [B][N][D]
  uint16_t* gTbf  = (uint16_t*)alloc((size_t)Bn * Dd * Nn * 2);  // g^T   [B][D][N]
  uint16_t* ybf   = (uint16_t*)alloc((size_t)Bn * Nn * Dd * 2);  // attn out [B][N][D]
  uint16_t* wth   = (uint16_t*)alloc((size_t)Dd * Cc * 2);
  uint16_t* wph   = (uint16_t*)alloc((size_t)Dd * Cc * 2);
  uint16_t* wg    = (uint16_t*)alloc((size_t)Dd * Cc * 2);
  uint16_t* wout  = (uint16_t*)alloc((size_t)Cc * Dd * 2);
  float*    zpre  = (float*)alloc((size_t)Bn * Cc * Nn * 4);     // pre-BN [B][C][N]
  float*    bnsum = (float*)alloc((size_t)Cc * 4);
  float*    bnsq  = (float*)alloc((size_t)Cc * 4);

  const int wcnt = Dd * Cc;  // 32768
  cvt_bf16_kernel<<<(wcnt + 255) / 256, 256, 0, stream>>>(theta_w, wth, wcnt);
  cvt_bf16_kernel<<<(wcnt + 255) / 256, 256, 0, stream>>>(phi_w,   wph, wcnt);
  cvt_bf16_kernel<<<(wcnt + 255) / 256, 256, 0, stream>>>(g_w,     wg,  wcnt);
  cvt_bf16_kernel<<<(wcnt + 255) / 256, 256, 0, stream>>>(out_w,   wout, wcnt);

  transpose_cvt_x<<<dim3(Nn / 32, Cc / 32, Bn), dim3(32, 8), 0, stream>>>(x, xt);

  const float scale = 0.088388347648318447f;  // 1/sqrt(D=128), folded into theta
  proj_gemm_kernel<<<Bn * (Nn / 16), 256, 0, stream>>>(xt, wth, theta_b, scale, thbf);
  proj_gemm_kernel<<<Bn * (Nn / 16), 256, 0, stream>>>(xt, wph, phi_b, 1.0f, phbf);
  proj_gemm_kernel<<<Bn * (Nn / 16), 256, 0, stream>>>(xt, wg,  g_b,   1.0f, gybf);

  transpose_g<<<dim3(Dd / 32, Nn / 32, Bn), dim3(32, 8), 0, stream>>>(gybf, gTbf);

  flash_attn_kernel<<<Bn * (Nn / 64), 128, 0, stream>>>(thbf, phbf, gTbf, ybf);

  init_stats<<<1, 256, 0, stream>>>(bnsum, bnsq);
  outconv_kernel<<<dim3(Bn * (Nn / 16), 2), 256, 0, stream>>>(ybf, wout, out_b, zpre, bnsum, bnsq);

  bn_residual_kernel<<<((size_t)Bn * Cc * Nn) / 256, 256, 0, stream>>>(
      x, zpre, bnsum, bnsq, gamma, beta, out);
}
